// DetrAttention_1262720385290
// MI455X (gfx1250) — matrix operand order repaired and checked by reading, not compiled, not run on hardware
//
#include <hip/hip_runtime.h>
#include <hip/hip_bf16.h>

typedef __attribute__((ext_vector_type(16))) __bf16 v16bf;
typedef __attribute__((ext_vector_type(8)))  __bf16 v8bf;
typedef __attribute__((ext_vector_type(8)))  float  v8f;
typedef __attribute__((ext_vector_type(16))) float  v16f;
typedef __attribute__((ext_vector_type(4)))  unsigned int uint32x4;
typedef __attribute__((ext_vector_type(8)))  int int32x8;
typedef __attribute__((ext_vector_type(4)))  int int32x4;

#define EMB 256
#define TT  2048
#define NB  8
#define NH  8
#define DH  32
#define NROWS (NB*TT)                    // 16384
#define HEADELEMS ((size_t)NB*NH*TT*DH)  // 4194304 per Q/K/V buffer
#define WELEMS   ((size_t)EMB*EMB)       // 65536 per weight matrix
#define QK_SCALE 0.17677669529663687f

static __device__ __forceinline__ v8f wmma_bf16(v16bf a, v16bf b, v8f c) {
    return __builtin_amdgcn_wmma_f32_16x16x32_bf16(false, a, false, b, (short)0, c, false, false);
}

static __device__ __forceinline__ v16bf ld16_2x8(const __bf16* lo, const __bf16* hi) {
    v8bf a = *(const v8bf*)lo;
    v8bf b = *(const v8bf*)hi;
    v16bf r;
    #pragma unroll
    for (int i = 0; i < 8; ++i) { r[i] = a[i]; r[8+i] = b[i]; }
    return r;
}

static __device__ __forceinline__ v16bf cvt16(v16f f) {
    v16bf r;
    #pragma unroll
    for (int i = 0; i < 16; ++i) r[i] = (__bf16)f[i];
    return r;
}

// -------------------- Kernel 0: one-shot weight conversion f32 -> bf16 ----
// grid 64 x 256; each thread converts 16 elements. Segments: Wq|Wk|Wv|Wo.
__global__ __launch_bounds__(256) void cvtw_kernel(
    const float* __restrict__ Wq, const float* __restrict__ Wk,
    const float* __restrict__ Wv, const float* __restrict__ Wo,
    __bf16* __restrict__ Wb)
{
    const size_t idx = ((size_t)blockIdx.x*256 + threadIdx.x) * 16;
    const int    seg = (int)(idx >> 16);          // uniform per block
    const size_t off = idx & (WELEMS - 1);
    const float* src = (seg==0) ? Wq : (seg==1) ? Wk : (seg==2) ? Wv : Wo;
    *(v16bf*)(Wb + idx) = cvt16(*(const v16f*)(src + off));
}

// -------------------- Kernel 1: fused QKV projection --------------------
// grid (NROWS/16, EMB/128, 3), block 256 (8 waves). Wave -> one 16x16 tile.
__global__ __launch_bounds__(256) void qkv_kernel(
    const float* __restrict__ hidden, const float* __restrict__ obj,
    const __bf16* __restrict__ Wb,
    const float* __restrict__ bq, const float* __restrict__ bk,
    const float* __restrict__ bv,
    __bf16* __restrict__ Qo, __bf16* __restrict__ Ko, __bf16* __restrict__ Vo)
{
    __shared__ __align__(64) __bf16 sA[16*EMB];   // 8KB

    const int lane = threadIdx.x & 31;
    const int wid  = threadIdx.x >> 5;
    const int ln   = lane & 15;
    const int hi   = lane >> 4;
    const int z    = blockIdx.z;                 // 0=Q 1=K 2=V
    const int m0   = blockIdx.x * 16;            // rows over B*T
    const int n0   = blockIdx.y * 128 + wid*16;  // cols over E

    // ---- stage A tile (uniform branch on z) ----
    {
        const int row  = threadIdx.x >> 4;        // 0..15
        const int col0 = (threadIdx.x & 15) * 16; // 0..240
        v16f hv = *(const v16f*)(hidden + (size_t)(m0 + row)*EMB + col0);
        if (z < 2) {
            v16f ov = *(const v16f*)(obj + (size_t)(m0 + row)*EMB + col0);
            #pragma unroll
            for (int i = 0; i < 16; ++i) hv[i] += ov[i];
        }
        *(v16bf*)&sA[row*EMB + col0] = cvt16(hv);
    }
    __syncthreads();

    const float*  bias = (z==0) ? bq : (z==1) ? bk : bv;
    __bf16*       Out  = (z==0) ? Qo : (z==1) ? Ko : Vo;
    const __bf16* wp   = Wb + (size_t)z*WELEMS + (size_t)(n0 + ln)*EMB;

    v8f acc = {};
    #pragma unroll
    for (int kc = 0; kc < EMB; kc += 32) {
        v16bf a = ld16_2x8(&sA[ln*EMB + kc + hi*16], &sA[ln*EMB + kc + hi*16 + 8]);
        v16bf b = *(const v16bf*)(wp + kc + hi*16);
        acc = wmma_bf16(a, b, acc);
    }

    const float bval  = bias[n0 + ln];
    const float scale = (z==0) ? QK_SCALE : 1.0f;
    const int e = n0 + ln;
    const int h = e >> 5, d = e & 31;
    #pragma unroll
    for (int r = 0; r < 8; ++r) {
        int t  = m0 + r + hi*8;          // flat row over B*T
        int bb = t >> 11, tl = t & (TT-1);
        float v = (acc[r] + bval) * scale;
        Out[(((size_t)bb*NH + h)*TT + tl)*DH + d] = (__bf16)v;
    }
}

// -------------------- Kernel 2: flash attention --------------------
// grid (T/128, B*H), block 256 (8 waves). Wave owns 16 query rows.
// K tile staged via Tensor Data Mover (TDM) when available.
__global__ __launch_bounds__(256) void attn_kernel(
    const __bf16* __restrict__ Q, const __bf16* __restrict__ K,
    const __bf16* __restrict__ V, const float* __restrict__ mask,
    __bf16* __restrict__ ATT)
{
    __shared__ __align__(16) __bf16 sK[64*32];       // 4KB  row-major [kr][d]
    __shared__ __align__(16) __bf16 sVt[32*64];      // 4KB  transposed [d][kr]
    __shared__ __align__(16) __bf16 sP[8][16*64];    // 16KB per-wave P tile
    __shared__ __align__(16) float  sM[128*64];      // 32KB mask tile

    const int lane = threadIdx.x & 31;
    const int wid  = threadIdx.x >> 5;
    const int ln   = lane & 15;
    const int hi   = lane >> 4;

    const int bh = blockIdx.y;          // b*NH + h
    const int b  = bh >> 3;
    const int h  = bh & 7;
    const int q0 = blockIdx.x * 128;
    const int qr = q0 + wid*16;

    const __bf16* Qbh = Q + (size_t)bh * TT * DH;
    const __bf16* Kbh = K + (size_t)bh * TT * DH;
    const __bf16* Vbh = V + (size_t)bh * TT * DH;
    const float*  Mb  = mask + (size_t)b * TT * TT;

    const int kvrow = threadIdx.x >> 2;          // 0..63
    const int kvoff = (threadIdx.x & 3) * 8;     // 0,8,16,24
    const int mrow2 = threadIdx.x >> 1;          // 0..127
    const int mcol2 = (threadIdx.x & 1) * 32;    // 0 or 32
    const float* mrowp = Mb + (size_t)(q0 + mrow2)*TT + mcol2;

    // Q fragment (16x32, A layout); SCALING already folded in.
    const __bf16* qp = Qbh + (size_t)(qr + ln)*DH;
    v16bf aq = ld16_2x8(qp + hi*16, qp + hi*16 + 8);

    float mrow[8], lrow[8];
    v8f o0 = {}, o1 = {};
    #pragma unroll
    for (int r = 0; r < 8; ++r) { mrow[r] = -1e30f; lrow[r] = 0.0f; }

#if __has_builtin(__builtin_amdgcn_tensor_load_to_lds)
    const unsigned ldsK = (unsigned)(uintptr_t)&sK[0];   // LDS aperture low bits
#endif

    for (int j0 = 0; j0 < TT; j0 += 64) {
        // ---- K tile: TDM async DMA (wave 0), else manual fallback ----
#if __has_builtin(__builtin_amdgcn_tensor_load_to_lds)
        if (wid == 0) {
            unsigned long long ga = (unsigned long long)(uintptr_t)(Kbh + (size_t)j0*DH);
            uint32x4 g0;
            g0.x = 1u;                                        // count=1 valid D#
            g0.y = ldsK;                                      // lds_addr (bytes)
            g0.z = (unsigned)(ga & 0xFFFFFFFFu);              // global_addr lo
            g0.w = (unsigned)((ga >> 32) & 0x1FFFFFFu) | (2u << 30); // hi | type=2
            int32x8 g1;
            g1[0] = (1 << 16);            // wg_mask=0, data_size=1 (2 bytes)
            g1[1] = (int)(32u << 16);     // tensor_dim0[15:0]=32 in [63:48]
            g1[2] = (int)(2048u << 16);   // {tensor_dim1[15:0]=2048, tdim0 hi=0}
            g1[3] = (int)(32u << 16);     // {tile_dim0=32, tensor_dim1 hi=0}
            g1[4] = 64;                   // {tile_dim2=0, tile_dim1=64}
            g1[5] = 32;                   // tensor_dim0_stride = 32 elements
            g1[6] = 0;
            g1[7] = 0;
            int32x4 gz = {0, 0, 0, 0};
#if __clang_major__ >= 23
            int32x8 gz8 = {0,0,0,0,0,0,0,0};
            __builtin_amdgcn_tensor_load_to_lds(g0, g1, gz, gz, gz8, 0);
#else
            __builtin_amdgcn_tensor_load_to_lds(g0, g1, gz, gz, 0);
#endif
#if __has_builtin(__builtin_amdgcn_s_wait_tensorcnt)
            __builtin_amdgcn_s_wait_tensorcnt(0);
#else
            asm volatile("s_wait_tensorcnt 0x0" ::: "memory");
#endif
        }
#else
        {
            const __bf16* kp = Kbh + (size_t)(j0 + kvrow)*DH + kvoff;
            *(v8bf*)&sK[kvrow*32 + kvoff] = *(const v8bf*)kp;
        }
#endif
        // ---- cooperative staging: V transposed + mask tile ----
        {
            const __bf16* vp = Vbh + (size_t)(j0 + kvrow)*DH + kvoff;
            v8bf vx = *(const v8bf*)vp;
            #pragma unroll
            for (int i = 0; i < 8; ++i) sVt[(kvoff + i)*64 + kvrow] = vx[i];

            #pragma unroll
            for (int jj = 0; jj < 8; ++jj) {
                float4 mv = *(const float4*)(mrowp + j0 + jj*4);
                *(float4*)&sM[mrow2*64 + mcol2 + jj*4] = mv;
            }
            if (j0 + 64 < TT)   // prefetch next mask tile (global_prefetch_b8)
                __builtin_prefetch(mrowp + j0 + 64, 0, 1);
        }
        __syncthreads();

        // ---- S = Q K^T (4 column tiles of 16) ----
        v8f s[4];
        #pragma unroll
        for (int c = 0; c < 4; ++c) {
            v16bf bk = *(const v16bf*)&sK[(c*16 + ln)*32 + hi*16];
            v8f zero = {};
            s[c] = wmma_bf16(aq, bk, zero);
        }

        // ---- add attention mask from LDS tile ----
        #pragma unroll
        for (int c = 0; c < 4; ++c) {
            #pragma unroll
            for (int r = 0; r < 8; ++r)
                s[c][r] += sM[(wid*16 + r + hi*8)*64 + c*16 + ln];
        }

        // ---- online softmax (row = r + hi*8; reduce across 16-lane half) ----
        float corr[8];
        #pragma unroll
        for (int r = 0; r < 8; ++r) {
            float mx = fmaxf(fmaxf(s[0][r], s[1][r]), fmaxf(s[2][r], s[3][r]));
            #pragma unroll
            for (int sh = 8; sh >= 1; sh >>= 1) mx = fmaxf(mx, __shfl_xor(mx, sh, 32));
            float mnew = fmaxf(mrow[r], mx);
            corr[r] = __expf(mrow[r] - mnew);
            mrow[r] = mnew;
            float rs = 0.0f;
            #pragma unroll
            for (int c = 0; c < 4; ++c) {
                float p = __expf(s[c][r] - mnew);
                s[c][r] = p;
                rs += p;
            }
            #pragma unroll
            for (int sh = 8; sh >= 1; sh >>= 1) rs += __shfl_xor(rs, sh, 32);
            lrow[r] = lrow[r]*corr[r] + rs;
            o0[r] *= corr[r];
            o1[r] *= corr[r];
        }

        // ---- P (16x64 bf16) to this wave's LDS tile (DS ops wave-ordered) ----
        __bf16* P = sP[wid];
        #pragma unroll
        for (int c = 0; c < 4; ++c)
            #pragma unroll
            for (int r = 0; r < 8; ++r)
                P[(r + hi*8)*64 + c*16 + ln] = (__bf16)s[c][r];

        // ---- O += P @ V : two K chunks x two output column tiles ----
        #pragma unroll
        for (int kc = 0; kc < 2; ++kc) {
            const __bf16* pp = P + (size_t)ln*64 + kc*32;
            v16bf ap = ld16_2x8(pp + hi*16, pp + hi*16 + 8);
            #pragma unroll
            for (int ct = 0; ct < 2; ++ct) {
                v16bf bvf = *(const v16bf*)&sVt[(ct*16 + ln)*64 + kc*32 + hi*16];
                if (ct == 0) o0 = wmma_bf16(ap, bvf, o0);
                else         o1 = wmma_bf16(ap, bvf, o1);
            }
        }
        __syncthreads();
    }

    // normalize + store attn (bf16, [B,T,E] with e = h*32 + d)
    #pragma unroll
    for (int r = 0; r < 8; ++r) {
        int bt = b*TT + qr + r + hi*8;
        float inv = 1.0f / lrow[r];
        size_t base = (size_t)bt*EMB + h*DH;
        ATT[base + ln]      = (__bf16)(o0[r] * inv);
        ATT[base + 16 + ln] = (__bf16)(o1[r] * inv);
    }
}

// -------------------- Kernel 3: output projection --------------------
// grid (NROWS/16, EMB/128), block 256. out = attn @ Wo^T + bo (f32 out)
__global__ __launch_bounds__(256) void outproj_kernel(
    const __bf16* __restrict__ ATT, const __bf16* __restrict__ WoB,
    const float* __restrict__ bo, float* __restrict__ out)
{
    __shared__ __align__(64) __bf16 sA[16*EMB];   // 8KB

    const int lane = threadIdx.x & 31;
    const int wid  = threadIdx.x >> 5;
    const int ln   = lane & 15;
    const int hi   = lane >> 4;
    const int m0   = blockIdx.x * 16;
    const int n0   = blockIdx.y * 128 + wid*16;

    {
        const int row  = threadIdx.x >> 4;
        const int col0 = (threadIdx.x & 15) * 16;
        *(v16bf*)&sA[row*EMB + col0] = *(const v16bf*)(ATT + (size_t)(m0 + row)*EMB + col0);
    }
    __syncthreads();

    const __bf16* wp = WoB + (size_t)(n0 + ln)*EMB;

    v8f acc = {};
    #pragma unroll
    for (int kc = 0; kc < EMB; kc += 32) {
        v16bf a = ld16_2x8(&sA[ln*EMB + kc + hi*16], &sA[ln*EMB + kc + hi*16 + 8]);
        v16bf b = *(const v16bf*)(wp + kc + hi*16);
        acc = wmma_bf16(a, b, acc);
    }

    const float bias = bo[n0 + ln];
    #pragma unroll
    for (int r = 0; r < 8; ++r)
        out[(size_t)(m0 + r + hi*8)*EMB + n0 + ln] = acc[r] + bias;
}

extern "C" void kernel_launch(void* const* d_in, const int* in_sizes, int n_in,
                              void* d_out, int out_size, void* d_ws, size_t ws_size,
                              hipStream_t stream) {
    (void)in_sizes; (void)n_in; (void)out_size; (void)ws_size;
    const float* hidden = (const float*)d_in[0];
    const float* obj    = (const float*)d_in[1];
    const float* mask   = (const float*)d_in[2];
    const float* Wq = (const float*)d_in[3];
    const float* bq = (const float*)d_in[4];
    const float* Wk = (const float*)d_in[5];
    const float* bk = (const float*)d_in[6];
    const float* Wv = (const float*)d_in[7];
    const float* bv = (const float*)d_in[8];
    const float* Wo = (const float*)d_in[9];
    const float* bo = (const float*)d_in[10];
    float* out = (float*)d_out;

    // workspace: Q | K | V | ATT (4 x 8MB bf16) | Wb (4 x 128KB bf16) = 32.5MB
    __bf16* Qb  = (__bf16*)d_ws;
    __bf16* Kb  = Qb + HEADELEMS;
    __bf16* Vb  = Kb + HEADELEMS;
    __bf16* ATT = Vb + HEADELEMS;
    __bf16* Wb  = ATT + HEADELEMS;       // Wq|Wk|Wv|Wo, 65536 each

    cvtw_kernel<<<dim3(64), 256, 0, stream>>>(Wq, Wk, Wv, Wo, Wb);
    qkv_kernel<<<dim3(NROWS/16, EMB/128, 3), 256, 0, stream>>>(
        hidden, obj, Wb, bq, bk, bv, Qb, Kb, Vb);
    attn_kernel<<<dim3(TT/128, NB*NH), 256, 0, stream>>>(Qb, Kb, Vb, mask, ATT);
    outproj_kernel<<<dim3(NROWS/16, EMB/128), 256, 0, stream>>>(ATT, Wb + 3*WELEMS, bo, out);
}
